// MSDeformAttn_82102594830689
// MI455X (gfx1250) — matrix operand order repaired and checked by reading, not compiled, not run on hardware
//
#include <hip/hip_runtime.h>
#include <hip/hip_bf16.h>
#include <math.h>

// ---------------------------------------------------------------------------
// MSDeformAttn for gfx1250 (CDNA5, wave32, WMMA).
// GEMMs run on v_wmma_f32_16x16x32_f16 (f16 A/B, f32 accum).
// ---------------------------------------------------------------------------

typedef __attribute__((ext_vector_type(16))) _Float16 v16h;
typedef __attribute__((ext_vector_type(8)))  _Float16 v8h;
typedef __attribute__((ext_vector_type(8)))  float    v8f;

#define D_MODEL  256
#define N_HEADS  8
#define N_LEVELS 4
#define N_POINTS 4
#define D_HEAD   32

// ---------------------------------------------------------------------------
// f32 -> f16 elementwise convert
// ---------------------------------------------------------------------------
__global__ void cvt_f32_to_f16(const float* __restrict__ src,
                               _Float16* __restrict__ dst, int n) {
    int i = blockIdx.x * blockDim.x + threadIdx.x;
    if (i < n) dst[i] = (_Float16)src[i];
}

// ---------------------------------------------------------------------------
// Wave-level WMMA GEMM:  out[m][n] = sum_k A[m][k] * W[n][k] + bias[n]
//   A : [M, 256] f16 row-major
//   W : [N, 256] f16 row-major  (note: W.T is the "B" matrix)
// Each wave computes a 16(M) x 64(N) strip: 4 f32 accumulators, 32 WMMAs.
//
// Fragment layouts used here (contiguous k order, 16-bit, wave32):
//   A 16x32 : lane L -> m = L&15, kb = (L>>4)*16 ;
//             halfs[0..7]  = A[m][k0+kb .. k0+kb+7]
//             halfs[8..15] = A[m][k0+kb+8 .. k0+kb+15]
//   B 32x16 : lane L -> n = L&15, kg = L>>4 ;
//             halfs[j] = B[k0 + kg*16 + j][n] = W[n][k0 + kg*16 + j]  (contig!)
//   C 16x16 : lane L -> col n = L&15, rows (L>>4)*8 + r,  r = 0..7
// ---------------------------------------------------------------------------
template <int N>
__global__ void gemm_wmma_f16(const _Float16* __restrict__ A,
                              const _Float16* __restrict__ W,
                              const float*    __restrict__ bias,
                              float*          __restrict__ outF32,  // may be null
                              _Float16*       __restrict__ outF16,  // may be null
                              int M) {
    constexpr int K  = 256;
    constexpr int NG = N / 64;                 // 16x64 strips per M-row

    const int wave   = blockIdx.x * (blockDim.x >> 5) + (threadIdx.x >> 5);
    const int mtiles = (M + 15) >> 4;
    if (wave >= mtiles * NG) return;           // wave-uniform exit (EXEC stays full)

    const int mtile = wave / NG;
    const int ng    = wave % NG;
    const int lane  = threadIdx.x & 31;
    const int lg    = lane >> 4;               // half-wave group: 0 or 1
    const int nloc  = lane & 15;

    int mrowA = mtile * 16 + (lane & 15);
    if (mrowA >= M) mrowA = M - 1;             // clamp loads; stores guarded below
    const _Float16* aptr = A + (size_t)mrowA * K + lg * 16;

    union AFrag { v16h v; v8h h[2]; };

    v8f acc[4] = {};
#pragma unroll
    for (int kc = 0; kc < K / 32; ++kc) {
        AFrag af;
        af.h[0] = *(const v8h*)(aptr + kc * 32);        // K = k0+kb   .. +7
        af.h[1] = *(const v8h*)(aptr + kc * 32 + 8);    // K = k0+kb+8 .. +15
#pragma unroll
        for (int j = 0; j < 4; ++j) {
            const int n = ng * 64 + j * 16 + nloc;
            const v16h b = *(const v16h*)(W + (size_t)n * K + kc * 32 + lg * 16);
            acc[j] = __builtin_amdgcn_wmma_f32_16x16x32_f16(
                false, af.v, false, b, (short)0, acc[j], false, false);
        }
    }

    const int rbase = mtile * 16 + lg * 8;
#pragma unroll
    for (int j = 0; j < 4; ++j) {
        const int n  = ng * 64 + j * 16 + nloc;
        const float bv = bias ? bias[n] : 0.0f;
#pragma unroll
        for (int r = 0; r < 8; ++r) {
            const int mrow = rbase + r;
            if (mrow < M) {
                const float val = acc[j][r] + bv;
                if (outF32) outF32[(size_t)mrow * N + n] = val;
                if (outF16) outF16[(size_t)mrow * N + n] = (_Float16)val;
            }
        }
    }
}

// ---------------------------------------------------------------------------
// Softmax over groups of 16 (per query, per head). One thread per (q, h).
// ---------------------------------------------------------------------------
__global__ void softmax16_kernel(float* __restrict__ logits, int nrows) {
    int i = blockIdx.x * blockDim.x + threadIdx.x;
    if (i >= nrows) return;
    float* p = logits + (size_t)i * 16;
    float m = p[0];
#pragma unroll
    for (int j = 1; j < 16; ++j) m = fmaxf(m, p[j]);
    float e[16];
    float s = 0.0f;
#pragma unroll
    for (int j = 0; j < 16; ++j) { e[j] = __expf(p[j] - m); s += e[j]; }
    const float inv = 1.0f / s;
#pragma unroll
    for (int j = 0; j < 16; ++j) p[j] = e[j] * inv;
}

// ---------------------------------------------------------------------------
// Deformable sampling core. One block (256 threads) per query.
// Thread t -> head h = t>>5, channel c = t&31. 32 lanes of one head read
// contiguous channels -> every gather is a coalesced 128B transaction.
// ---------------------------------------------------------------------------
__device__ __forceinline__ float fetch_val(const float* __restrict__ v,
                                           int st, int hh, int ww,
                                           int ix, int iy, int ch) {
    if (ix < 0 || ix >= ww || iy < 0 || iy >= hh) return 0.0f;
    return v[(size_t)(st + iy * ww + ix) * D_MODEL + ch];
}

__global__ void msdeform_sample_kernel(const float* __restrict__ value,  // [Lq,256]
                                       const float* __restrict__ offs,   // [Lq,256]
                                       const float* __restrict__ attn,   // [Lq,128]
                                       const float* __restrict__ refp,   // [Lq,4,2]
                                       const int*   __restrict__ shapes, // [4,2] (H,W)
                                       _Float16*    __restrict__ ms16,   // [Lq,256]
                                       int Lq) {
    const int q = blockIdx.x;
    const int t = threadIdx.x;
    const int h = t >> 5;
    const int c = t & 31;

    __shared__ int sSt[N_LEVELS], sH[N_LEVELS], sW[N_LEVELS];
    if (t == 0) {
        int s = 0;
        for (int l = 0; l < N_LEVELS; ++l) {
            sSt[l] = s; sH[l] = shapes[2 * l]; sW[l] = shapes[2 * l + 1];
            s += sH[l] * sW[l];
        }
    }
    __syncthreads();

    const float* offq = offs + (size_t)q * (N_HEADS * N_LEVELS * N_POINTS * 2);
    const float* attq = attn + (size_t)q * (N_HEADS * N_LEVELS * N_POINTS);
    const float* refq = refp + (size_t)q * (N_LEVELS * 2);
    const int    ch   = h * D_HEAD + c;

    float acc = 0.0f;
#pragma unroll
    for (int l = 0; l < N_LEVELS; ++l) {
        const int hh = sH[l], ww = sW[l], st = sSt[l];
        const float rx = refq[l * 2 + 0];
        const float ry = refq[l * 2 + 1];
        const float invW = 1.0f / (float)ww;
        const float invH = 1.0f / (float)hh;
#pragma unroll
        for (int p = 0; p < N_POINTS; ++p) {
            const int oi = ((h * N_LEVELS + l) * N_POINTS + p) * 2;
            const float lx = rx + offq[oi + 0] * invW;
            const float ly = ry + offq[oi + 1] * invH;
            const float x = lx * (float)ww - 0.5f;
            const float y = ly * (float)hh - 0.5f;
            const float x0f = floorf(x), y0f = floorf(y);
            const float wx = x - x0f, wy = y - y0f;
            const int x0 = (int)x0f, y0 = (int)y0f;

            const float v00 = fetch_val(value, st, hh, ww, x0,     y0,     ch);
            const float v10 = fetch_val(value, st, hh, ww, x0 + 1, y0,     ch);
            const float v01 = fetch_val(value, st, hh, ww, x0,     y0 + 1, ch);
            const float v11 = fetch_val(value, st, hh, ww, x0 + 1, y0 + 1, ch);

            const float bil = v00 * (1.0f - wx) * (1.0f - wy)
                            + v10 * wx          * (1.0f - wy)
                            + v01 * (1.0f - wx) * wy
                            + v11 * wx          * wy;
            acc += attq[h * (N_LEVELS * N_POINTS) + l * N_POINTS + p] * bil;
        }
    }
    ms16[(size_t)q * D_MODEL + ch] = (_Float16)acc;
}

// ---------------------------------------------------------------------------
// Host driver
// ---------------------------------------------------------------------------
static inline size_t align256(size_t x) { return (x + 255) & ~(size_t)255; }

extern "C" void kernel_launch(void* const* d_in, const int* in_sizes, int n_in,
                              void* d_out, int out_size, void* d_ws, size_t ws_size,
                              hipStream_t stream) {
    const float* query  = (const float*)d_in[0];   // [Lq,256]
    const float* refp   = (const float*)d_in[1];   // [Lq,4,2]
    const float* inflat = (const float*)d_in[2];   // [Lq,256]
    const int*   shapes = (const int*)  d_in[3];   // [4,2]
    const float* W_off  = (const float*)d_in[4];   // [256,256]
    const float* b_off  = (const float*)d_in[5];   // [256]
    const float* W_attn = (const float*)d_in[6];   // [128,256]
    const float* b_attn = (const float*)d_in[7];   // [128]
    const float* W_val  = (const float*)d_in[8];   // [256,256]
    const float* b_val  = (const float*)d_in[9];   // [256]
    const float* W_out  = (const float*)d_in[10];  // [256,256]
    const float* b_out  = (const float*)d_in[11];  // [256]
    float* out = (float*)d_out;

    const int Lq   = in_sizes[0] / D_MODEL;        // 19947
    const int n_op = N_HEADS * N_LEVELS * N_POINTS; // 128

    // ---- workspace carve-up ----
    char* ws = (char*)d_ws;
    size_t off = 0;
    _Float16* q16   = (_Float16*)(ws + off); off = align256(off + (size_t)Lq * D_MODEL * 2);
    _Float16* in16  = (_Float16*)(ws + off); off = align256(off + (size_t)Lq * D_MODEL * 2);
    _Float16* Wv16  = (_Float16*)(ws + off); off = align256(off + (size_t)D_MODEL * D_MODEL * 2);
    _Float16* Wo16  = (_Float16*)(ws + off); off = align256(off + (size_t)(n_op * 2) * D_MODEL * 2);
    _Float16* Wa16  = (_Float16*)(ws + off); off = align256(off + (size_t)n_op * D_MODEL * 2);
    _Float16* Wout16= (_Float16*)(ws + off); off = align256(off + (size_t)D_MODEL * D_MODEL * 2);
    float*    valF  = (float*)   (ws + off); off = align256(off + (size_t)Lq * D_MODEL * 4);
    float*    offF  = (float*)   (ws + off); off = align256(off + (size_t)Lq * (n_op * 2) * 4);
    float*    attF  = (float*)   (ws + off); off = align256(off + (size_t)Lq * n_op * 4);
    _Float16* ms16  = (_Float16*)(ws + off); off = align256(off + (size_t)Lq * D_MODEL * 2);
    (void)ws_size; (void)n_in; (void)out_size;

    // ---- 1) f32 -> f16 conversions ----
    const int cb = 256;
    {
        int n;
        n = Lq * D_MODEL;
        cvt_f32_to_f16<<<(n + cb - 1) / cb, cb, 0, stream>>>(query,  q16,  n);
        cvt_f32_to_f16<<<(n + cb - 1) / cb, cb, 0, stream>>>(inflat, in16, n);
        n = D_MODEL * D_MODEL;
        cvt_f32_to_f16<<<(n + cb - 1) / cb, cb, 0, stream>>>(W_val, Wv16,  n);
        cvt_f32_to_f16<<<(n + cb - 1) / cb, cb, 0, stream>>>(W_off, Wo16,  n);
        cvt_f32_to_f16<<<(n + cb - 1) / cb, cb, 0, stream>>>(W_out, Wout16,n);
        n = n_op * D_MODEL;
        cvt_f32_to_f16<<<(n + cb - 1) / cb, cb, 0, stream>>>(W_attn, Wa16, n);
    }

    // ---- 2) input GEMMs (WMMA) ----
    const int mtiles = (Lq + 15) / 16;
    {
        // value = inflat @ W_val.T + b_val        [Lq, 256] f32
        int waves = mtiles * (256 / 64);
        int blocks = (waves + 7) / 8;
        gemm_wmma_f16<256><<<blocks, 256, 0, stream>>>(in16, Wv16, b_val, valF, nullptr, Lq);
        // off = query @ W_off.T + b_off           [Lq, 256] f32
        gemm_wmma_f16<256><<<blocks, 256, 0, stream>>>(q16, Wo16, b_off, offF, nullptr, Lq);
        // attn logits = query @ W_attn.T + b_attn [Lq, 128] f32
        int waves128 = mtiles * (128 / 64);
        int blocks128 = (waves128 + 7) / 8;
        gemm_wmma_f16<128><<<blocks128, 256, 0, stream>>>(q16, Wa16, b_attn, attF, nullptr, Lq);
    }

    // ---- 3) softmax over 16 per (q, head) ----
    {
        int nrows = Lq * N_HEADS;
        softmax16_kernel<<<(nrows + cb - 1) / cb, cb, 0, stream>>>(attF, nrows);
    }

    // ---- 4) deformable bilinear sampling -> ms16 ----
    msdeform_sample_kernel<<<Lq, 256, 0, stream>>>(valF, offF, attF, refp, shapes, ms16, Lq);

    // ---- 5) output GEMM (WMMA) straight into d_out ----
    {
        int waves = mtiles * (256 / 64);
        int blocks = (waves + 7) / 8;
        gemm_wmma_f16<256><<<blocks, 256, 0, stream>>>(ms16, Wout16, b_out, out, nullptr, Lq);
    }
}
